// GAT_GCN_24507083391730
// MI455X (gfx1250) — compile-verified, matrix operand order checked by reading, not run on hardware
//
#include <hip/hip_runtime.h>
#include <stdint.h>

typedef unsigned short u16;
typedef unsigned int   u32;

typedef __attribute__((ext_vector_type(16))) __bf16 v16bf;
typedef __attribute__((ext_vector_type(8)))  float  v8f;

union FragU { u32 u[8]; v16bf v; };

#define HEADS 10
#define HDIM  780
#define FXD   78

// ---------- small helpers ----------
__device__ __forceinline__ u16 f2bf(float f) {
  u32 u = __float_as_uint(f);
  u += 0x7FFFu + ((u >> 16) & 1u);           // round-to-nearest-even
  return (u16)(u >> 16);
}
__device__ __forceinline__ float bf2f(u16 h) { return __uint_as_float(((u32)h) << 16); }

__device__ __forceinline__ v8f vzero8() {
  v8f z;
#pragma unroll
  for (int i = 0; i < 8; ++i) z[i] = 0.0f;
  return z;
}

__device__ __forceinline__ v8f wmma_bf16(v16bf a, v16bf b, v8f c) {
  return __builtin_amdgcn_wmma_f32_16x16x32_bf16(false, a, false, b, (short)0, c, false, false);
}

__device__ __forceinline__ void atomicMaxF32(float* addr, float val) {
  u32* ua = (u32*)addr;
  u32 cur = *ua;
  while (__uint_as_float(cur) < val) {
    u32 prev = atomicCAS(ua, cur, __float_as_uint(val));
    if (prev == cur) break;
    cur = prev;
  }
}

// A/B fragment (16x32 bf16) per ISA 7.12.2.  Fragment = two contiguous 16B
// runs per lane -> two global_load_b128, no predication (buffers are padded).
__device__ __forceinline__ v16bf load_frag(const u16* __restrict__ base, int row0,
                                           int ld, int kc) {
  const int lane = threadIdx.x & 31;
  const u16* p = base + (size_t)(row0 + (lane & 15)) * ld + kc + ((lane >> 4) * 8);
  uint4 lo = *(const uint4*)p;          // k halves +0..7
  uint4 hi = *(const uint4*)(p + 16);   // k halves +16..23
  FragU f;
  f.u[0] = lo.x; f.u[1] = lo.y; f.u[2] = lo.z; f.u[3] = lo.w;
  f.u[4] = hi.x; f.u[5] = hi.y; f.u[6] = hi.z; f.u[7] = hi.w;
  return f.v;
}

// im2col A fragment for conv1d: k = kk*Cin + ci, input [Lpad+15, Cin] bf16
// (zero-padded).  Each 8-half run stays inside one kk row (ci % 8 == 0).
__device__ __forceinline__ v16bf load_frag_conv(const u16* __restrict__ inb, int row0,
                                                int Cin, int cinLog2, int kc) {
  const int lane = threadIdx.x & 31;
  const int l    = row0 + (lane & 15);
  const int k0   = kc + ((lane >> 4) * 8);
  const int k1   = k0 + 16;
  const int kk0 = k0 >> cinLog2, ci0 = k0 & (Cin - 1);
  const int kk1 = k1 >> cinLog2, ci1 = k1 & (Cin - 1);
  uint4 lo = *(const uint4*)(inb + (size_t)(l + kk0) * Cin + ci0);
  uint4 hi = *(const uint4*)(inb + (size_t)(l + kk1) * Cin + ci1);
  FragU f;
  f.u[0] = lo.x; f.u[1] = lo.y; f.u[2] = lo.z; f.u[3] = lo.w;
  f.u[4] = hi.x; f.u[5] = hi.y; f.u[6] = hi.z; f.u[7] = hi.w;
  return f.v;
}

// C tile store per ISA: row = vgpr + 8*(lane>=16), col = lane&15.
__device__ __forceinline__ void store_tile(float* Cf, u16* Cb, int ldc,
                                           int row0, int col0, int M, int N, v8f acc,
                                           const float* scale, const float* shift, int relu) {
  const int lane  = threadIdx.x & 31;
  const int col   = col0 + (lane & 15);
  const int rbase = row0 + ((lane >> 4) ? 8 : 0);
  if (col >= N) return;
  const float sc = scale ? scale[col] : 1.0f;
  const float sh = shift ? shift[col] : 0.0f;
#pragma unroll
  for (int r = 0; r < 8; ++r) {
    int rr = rbase + r;
    if (rr < M) {
      float v = acc[r] * sc + sh;
      if (relu) v = fmaxf(v, 0.0f);
      if (Cf) Cf[(size_t)rr * ldc + col] = v;
      if (Cb) Cb[(size_t)rr * ldc + col] = f2bf(v);
    }
  }
}

// ---------- WMMA GEMM: C[M,N] = A[M,Kp](bf16) x Bt[N,Kp](bf16)^T ----------
// A/Bt padded: rows to block tile, Kp multiple of 32 (pad zero-filled).
__global__ void __launch_bounds__(128)
k_gemm_bf16(const u16* __restrict__ A, const u16* __restrict__ Bt,
            float* Cf, u16* Cb, int M, int N, int Kp, int ldc,
            const float* scale, const float* shift, int relu) {
  const int wave = threadIdx.x >> 5;
  const int row0 = blockIdx.y * 64 + (wave >> 1) * 32;
  const int col0 = blockIdx.x * 64 + (wave & 1) * 32;
  v8f a00 = vzero8(), a01 = vzero8(), a10 = vzero8(), a11 = vzero8();
  for (int kc = 0; kc < Kp; kc += 32) {
    v16bf af0 = load_frag(A,  row0,      Kp, kc);
    v16bf af1 = load_frag(A,  row0 + 16, Kp, kc);
    v16bf bf0 = load_frag(Bt, col0,      Kp, kc);
    v16bf bf1 = load_frag(Bt, col0 + 16, Kp, kc);
    a00 = wmma_bf16(af0, bf0, a00);
    a01 = wmma_bf16(af0, bf1, a01);
    a10 = wmma_bf16(af1, bf0, a10);
    a11 = wmma_bf16(af1, bf1, a11);
  }
  store_tile(Cf, Cb, ldc, row0,      col0,      M, N, a00, scale, shift, relu);
  store_tile(Cf, Cb, ldc, row0,      col0 + 16, M, N, a01, scale, shift, relu);
  store_tile(Cf, Cb, ldc, row0 + 16, col0,      M, N, a10, scale, shift, relu);
  store_tile(Cf, Cb, ldc, row0 + 16, col0 + 16, M, N, a11, scale, shift, relu);
}

// ---------- conv1d (VALID, ks=16) as implicit GEMM, fused BN+ReLU ----------
__global__ void __launch_bounds__(128)
k_conv_wmma(const u16* __restrict__ in, const u16* __restrict__ wt, u16* out,
            int Lout, int cinLog2, int Cout,
            long long inBatchStride, long long outBatchStride,
            const float* scale, const float* shift) {
  const int Cin  = 1 << cinLog2;
  const int Ktot = Cin << 4;
  const int b    = blockIdx.z;
  const u16* inb = in  + (size_t)b * inBatchStride;
  u16* outb      = out + (size_t)b * outBatchStride;
  const int wave = threadIdx.x >> 5;
  const int row0 = blockIdx.y * 64 + (wave >> 1) * 32;
  const int col0 = blockIdx.x * 64 + (wave & 1) * 32;
  v8f a00 = vzero8(), a01 = vzero8(), a10 = vzero8(), a11 = vzero8();
  for (int kc = 0; kc < Ktot; kc += 32) {
    v16bf af0 = load_frag_conv(inb, row0,      Cin, cinLog2, kc);
    v16bf af1 = load_frag_conv(inb, row0 + 16, Cin, cinLog2, kc);
    v16bf bf0 = load_frag(wt, col0,      Ktot, kc);
    v16bf bf1 = load_frag(wt, col0 + 16, Ktot, kc);
    a00 = wmma_bf16(af0, bf0, a00);
    a01 = wmma_bf16(af0, bf1, a01);
    a10 = wmma_bf16(af1, bf0, a10);
    a11 = wmma_bf16(af1, bf1, a11);
  }
  store_tile(nullptr, outb, Cout, row0,      col0,      Lout, Cout, a00, scale, shift, 1);
  store_tile(nullptr, outb, Cout, row0,      col0 + 16, Lout, Cout, a01, scale, shift, 1);
  store_tile(nullptr, outb, Cout, row0 + 16, col0,      Lout, Cout, a10, scale, shift, 1);
  store_tile(nullptr, outb, Cout, row0 + 16, col0 + 16, Lout, Cout, a11, scale, shift, 1);
}

// ---------- elementwise / prep kernels ----------
__global__ void k_fill_f32(float* p, float v, int n) {
  int i = blockIdx.x * blockDim.x + threadIdx.x;
  if (i < n) p[i] = v;
}
__global__ void k_fill_u32(u32* p, u32 v, long long n) {
  long long i = (long long)blockIdx.x * blockDim.x + threadIdx.x;
  if (i < n) p[i] = v;
}
// f32 [rows,cols] -> bf16 [rows, ldd] (pad pre-zeroed)
__global__ void k_copy_pad_bf16(const float* s, u16* d, int rows, int cols, int ldd) {
  int i = blockIdx.x * blockDim.x + threadIdx.x;
  if (i >= rows * cols) return;
  int r = i / cols, c = i % cols;
  d[(size_t)r * ldd + c] = f2bf(s[i]);
}
// W[K,N] f32 -> Wt[N, ldd] bf16 (transposed, pad pre-zeroed)
__global__ void k_transpose_bf16(const float* W, u16* Wt, int K, int N, int ldd) {
  int i = blockIdx.x * blockDim.x + threadIdx.x;
  if (i >= K * N) return;
  int n = i / K, k = i % K;
  Wt[(size_t)n * ldd + k] = f2bf(W[(size_t)k * N + n]);
}
// w[Co,Ci,16] -> wt[Co, kk*Ci+ci] (pad rows pre-zeroed)
__global__ void k_conv_wreorder(const float* w, u16* wt, int Co, int Ci) {
  int i = blockIdx.x * blockDim.x + threadIdx.x;
  if (i >= Co * Ci * 16) return;
  int co = i / (Ci * 16), r = i % (Ci * 16), ci = r / 16, kk = r % 16;
  wt[(size_t)co * Ci * 16 + kk * Ci + ci] = f2bf(w[i]);
}
__global__ void k_bn_fold(const float* cb, const float* g, const float* bb,
                          const float* m, const float* v, float* sc, float* sh, int C) {
  int i = blockIdx.x * blockDim.x + threadIdx.x;
  if (i >= C) return;
  float s = g[i] * rsqrtf(v[i] + 1e-5f);
  sc[i] = s;
  sh[i] = (cb[i] - m[i]) * s + bb[i];
}

// ---------- GAT / GCN graph kernels ----------
__device__ __forceinline__ void edge_sd(const int* ei, int e, int NE, int& s, int& d) {
  if (e < NE) { s = ei[e]; d = ei[NE + e]; } else { s = e - NE; d = e - NE; }
}

__global__ void k_gat_scores(const float* h0, const float* asrc, const float* adst,
                             float* as_, float* ad_, int NN) {
  int i = blockIdx.x * blockDim.x + threadIdx.x;
  if (i >= NN * HEADS) return;
  int n = i / HEADS, hd = i % HEADS;
  const float* hp = h0 + (size_t)n * HDIM + hd * FXD;
  float s = 0.f, t = 0.f;
  for (int f = 0; f < FXD; ++f) {
    float hv = hp[f];
    s += hv * asrc[hd * FXD + f];
    t += hv * adst[hd * FXD + f];
  }
  as_[i] = s; ad_[i] = t;
}
__global__ void k_edge_max(const int* ei, const float* as_, const float* ad_,
                           float* mx, int NE, int ET) {
  int i = blockIdx.x * blockDim.x + threadIdx.x;
  if (i >= ET * HEADS) return;
  int e = i / HEADS, hd = i % HEADS, s, d;
  edge_sd(ei, e, NE, s, d);
  float ev = as_[s * HEADS + hd] + ad_[d * HEADS + hd];
  ev = ev > 0.f ? ev : 0.2f * ev;
  atomicMaxF32(&mx[d * HEADS + hd], ev);
}
__global__ void k_edge_p(const int* ei, const float* as_, const float* ad_,
                         const float* mx, float* pb, float* z, int NE, int ET) {
  int i = blockIdx.x * blockDim.x + threadIdx.x;
  if (i >= ET * HEADS) return;
  int e = i / HEADS, hd = i % HEADS, s, d;
  edge_sd(ei, e, NE, s, d);
  float ev = as_[s * HEADS + hd] + ad_[d * HEADS + hd];
  ev = ev > 0.f ? ev : 0.2f * ev;
  float pv = __expf(ev - mx[d * HEADS + hd]);
  pb[i] = pv;
  atomicAdd(&z[d * HEADS + hd], pv);
}
__global__ void k_deg(const int* ei, float* deg, int NE, int ET) {
  int e = blockIdx.x * blockDim.x + threadIdx.x;
  if (e >= ET) return;
  int s, d; edge_sd(ei, e, NE, s, d);
  atomicAdd(&deg[d], 1.0f);
}
__global__ void k_edge_aggr_gat(const int* ei, const float* h0, const float* pb,
                                const float* z, float* h1, int NE) {
  int e = blockIdx.x, s, d;
  edge_sd(ei, e, NE, s, d);
  __shared__ float sal[HEADS];
  if (threadIdx.x < HEADS)
    sal[threadIdx.x] = pb[(size_t)e * HEADS + threadIdx.x] / z[(size_t)d * HEADS + threadIdx.x];
  __syncthreads();
  const float* hs = h0 + (size_t)s * HDIM;
  float* hd = h1 + (size_t)d * HDIM;
  for (int f = threadIdx.x; f < HDIM; f += blockDim.x)
    atomicAdd(&hd[f], hs[f] * sal[f / FXD]);
}
// h1[NN,780] f32 -> hb[NN, ld] bf16 with bias+relu (pad pre-zeroed)
__global__ void k_gat_finish(const float* h1, const float* gb, u16* hb, int NN, int ld) {
  int i = blockIdx.x * blockDim.x + threadIdx.x;
  if (i >= NN * HDIM) return;
  int n = i / HDIM, f = i % HDIM;
  float v = fmaxf(h1[i] + gb[f], 0.0f);
  hb[(size_t)n * ld + f] = f2bf(v);
}
__global__ void k_edge_aggr_gcn(const int* ei, const float* hw, const float* deg,
                                float* h2, int NE) {
  int e = blockIdx.x, s, d;
  edge_sd(ei, e, NE, s, d);
  float norm = rsqrtf(deg[s]) * rsqrtf(deg[d]);
  const float* hs = hw + (size_t)s * HDIM;
  float* hd = h2 + (size_t)d * HDIM;
  for (int f = threadIdx.x; f < HDIM; f += blockDim.x)
    atomicAdd(&hd[f], hs[f] * norm);
}
__global__ void k_gcn_finish(float* h2, const float* gb, int total) {
  int i = blockIdx.x * blockDim.x + threadIdx.x;
  if (i >= total) return;
  h2[i] = fmaxf(h2[i] + gb[i % HDIM], 0.0f);
}
__global__ void k_pool(const float* h2, const int* batch, float* gmx, float* gsm,
                       float* cnt, int NN) {
  int i = blockIdx.x * blockDim.x + threadIdx.x;
  if (i >= NN * HDIM) return;
  int n = i / HDIM, f = i % HDIM;
  int g = batch[n];
  float v = h2[i];
  atomicMaxF32(&gmx[(size_t)g * HDIM + f], v);
  atomicAdd(&gsm[(size_t)g * HDIM + f], v);
  if (f == 0) atomicAdd(&cnt[g], 1.0f);
}
__global__ void k_pool_finish(const float* gmx, const float* gsm, const float* cnt,
                              u16* gcat, int NG, int ld) {
  int i = blockIdx.x * blockDim.x + threadIdx.x;
  if (i >= NG * HDIM) return;
  int g = i / HDIM, f = i % HDIM;
  float mx = gmx[i];
  if (mx < -1e29f) mx = 0.0f;                 // empty-graph segment_max -> 0
  float mean = gsm[i] / fmaxf(cnt[g], 1.0f);
  gcat[(size_t)g * ld + f]        = f2bf(mx);
  gcat[(size_t)g * ld + HDIM + f] = f2bf(mean);
}

// ---------- protein branch helpers ----------
__global__ void k_embed(const int* target, const float* emb, u16* xt,
                        int total, int lrows) {
  int i = blockIdx.x * blockDim.x + threadIdx.x;
  if (i >= total) return;
  int c = i % 128, r = i / 128, l = r % 1000, b = r / 1000;
  int t = target[b * 1000 + l];
  xt[((size_t)b * lrows + l) * 128 + c] = f2bf(emb[t * 128 + c]);
}
__global__ void k_maxpool_conv(const u16* y, u16* pooled, int Lout, int lrows,
                               int C, int NG) {
  int i = blockIdx.x * blockDim.x + threadIdx.x;
  if (i >= NG * C) return;
  int b = i / C, c = i % C;
  float v = -1e30f;
  for (int l = 0; l < Lout; ++l)
    v = fmaxf(v, bf2f(y[((size_t)b * lrows + l) * C + c]));
  pooled[i] = f2bf(v);
}
__global__ void k_out_dot(const u16* a2, const float* ow, const float* ob,
                          float* out, int NG) {
  int b = blockIdx.x * blockDim.x + threadIdx.x;
  if (b >= NG) return;
  float acc = ob[0];
  for (int j = 0; j < 512; ++j) acc += bf2f(a2[b * 512 + j]) * ow[j];
  out[b] = acc;
}

// =======================================================================
extern "C" void kernel_launch(void* const* d_in, const int* in_sizes, int n_in,
                              void* d_out, int out_size, void* d_ws, size_t ws_size,
                              hipStream_t stream) {
  (void)in_sizes; (void)n_in; (void)out_size; (void)ws_size;
  const int NN = 100000, NE = 800000, ET = NE + NN, NG = 512, SEQ = 1000;
  const int L1 = 985, C1 = 32;
  const int L2 = 970, C2 = 64;
  const int L3 = 955, C3 = 96;
  // padded dims
  const int NN_P = 100032;          // roundup64(NN)
  const int KX   = 96;              // roundup32(78)
  const int HD_P = 800;             // roundup32(780)
  const int NW1  = 832;             // roundup64(780) rows for gat_wt/gcn_wt
  const int KG1  = 1568;            // roundup32(1560)
  const int NF1  = 1536;            // roundup64(1500)
  const int KG2  = 1504;            // roundup32(1500)
  const int LROW = 1040;            // roundup64(Lout)+15 halo (conv inputs)
  const int Y3R  = 960;

  const float* x      = (const float*)d_in[0];
  const int*   ei     = (const int*)d_in[1];
  const int*   batch  = (const int*)d_in[2];
  const int*   target = (const int*)d_in[3];
  const float* gat_w  = (const float*)d_in[4];
  const float* asrc   = (const float*)d_in[5];
  const float* adst   = (const float*)d_in[6];
  const float* gat_b  = (const float*)d_in[7];
  const float* gcn_w  = (const float*)d_in[8];
  const float* gcn_b  = (const float*)d_in[9];
  const float* fcg1_w = (const float*)d_in[10];
  const float* fcg1_b = (const float*)d_in[11];
  const float* fcg2_w = (const float*)d_in[12];
  const float* fcg2_b = (const float*)d_in[13];
  const float* emb    = (const float*)d_in[14];
  const float* c1_w = (const float*)d_in[15], *c1_b = (const float*)d_in[16];
  const float* bn1g = (const float*)d_in[17], *bn1b = (const float*)d_in[18],
             * bn1m = (const float*)d_in[19], *bn1v = (const float*)d_in[20];
  const float* c2_w = (const float*)d_in[21], *c2_b = (const float*)d_in[22];
  const float* bn2g = (const float*)d_in[23], *bn2b = (const float*)d_in[24],
             * bn2m = (const float*)d_in[25], *bn2v = (const float*)d_in[26];
  const float* c3_w = (const float*)d_in[27], *c3_b = (const float*)d_in[28];
  const float* bn3g = (const float*)d_in[29], *bn3b = (const float*)d_in[30],
             * bn3m = (const float*)d_in[31], *bn3v = (const float*)d_in[32];
  const float* fcxt_w = (const float*)d_in[33], *fcxt_b = (const float*)d_in[34];
  const float* bnfg = (const float*)d_in[35], *bnfb = (const float*)d_in[36],
             * bnfm = (const float*)d_in[37], *bnfv = (const float*)d_in[38];
  const float* fc1_w = (const float*)d_in[39], *fc1_b = (const float*)d_in[40];
  const float* fc2_w = (const float*)d_in[41], *fc2_b = (const float*)d_in[42];
  const float* out_w = (const float*)d_in[43], *out_b = (const float*)d_in[44];

  // ---- bump allocator in d_ws ----
  char* base = (char*)d_ws; size_t off = 0;
  auto alloc = [&](size_t bytes) -> void* {
    void* p = base + off;
    off = (off + bytes + 255) & ~(size_t)255;
    return p;
  };
  float* bufP   = (float*)alloc((size_t)NN * HDIM * 4);     // h0 -> h2
  float* bufQ   = (float*)alloc((size_t)NN * HDIM * 4);     // h1 -> hw
  u16*   hb     = (u16*)  alloc((size_t)NN_P * HD_P * 2);
  u16*   xb     = (u16*)  alloc((size_t)NN_P * KX * 2);
  float* as_    = (float*)alloc((size_t)NN * HEADS * 4);
  float* ad_    = (float*)alloc((size_t)NN * HEADS * 4);
  float* mmax   = (float*)alloc((size_t)NN * HEADS * 4);
  float* zsum   = (float*)alloc((size_t)NN * HEADS * 4);
  float* pbuf   = (float*)alloc((size_t)ET * HEADS * 4);
  float* deg    = (float*)alloc((size_t)NN * 4);
  float* gmx    = (float*)alloc((size_t)NG * HDIM * 4);
  float* gsm    = (float*)alloc((size_t)NG * HDIM * 4);
  float* cnt    = (float*)alloc((size_t)NG * 4);
  u16*   gcat   = (u16*)alloc((size_t)NG * KG1 * 2);
  u16*   gg     = (u16*)alloc((size_t)NG * KG2 * 2);
  u16*   headin = (u16*)alloc((size_t)NG * 256 * 2);
  u16*   a1     = (u16*)alloc((size_t)NG * 1024 * 2);
  u16*   a2     = (u16*)alloc((size_t)NG * 512 * 2);
  u16*   gat_wt = (u16*)alloc((size_t)NW1 * KX * 2);
  u16*   gcn_wt = (u16*)alloc((size_t)NW1 * HD_P * 2);
  u16*   fcg1wt = (u16*)alloc((size_t)NF1 * KG1 * 2);
  u16*   fcg2wt = (u16*)alloc((size_t)128 * KG2 * 2);
  u16*   fcxtwt = (u16*)alloc((size_t)128 * 96 * 2);
  u16*   fc1wt  = (u16*)alloc((size_t)1024 * 256 * 2);
  u16*   fc2wt  = (u16*)alloc((size_t)512 * 1024 * 2);
  u16*   c1wt   = (u16*)alloc((size_t)64 * 2048 * 2);
  u16*   c2wt   = (u16*)alloc((size_t)64 * 512 * 2);
  u16*   c3wt   = (u16*)alloc((size_t)128 * 1024 * 2);
  float* s1 = (float*)alloc(C1 * 4); float* t1 = (float*)alloc(C1 * 4);
  float* s2 = (float*)alloc(C2 * 4); float* t2 = (float*)alloc(C2 * 4);
  float* s3 = (float*)alloc(C3 * 4); float* t3 = (float*)alloc(C3 * 4);
  float* sf = (float*)alloc(128 * 4); float* tf = (float*)alloc(128 * 4);
  u16*   xt0    = (u16*)alloc((size_t)NG * LROW * 128 * 2);
  u16*   y1     = (u16*)alloc((size_t)NG * LROW * C1 * 2);
  u16*   y2     = (u16*)alloc((size_t)NG * LROW * C2 * 2);
  u16*   y3     = (u16*)alloc((size_t)NG * Y3R * C3 * 2);
  u16*   pooled = (u16*)alloc((size_t)NG * 96 * 2);

  const int T = 256;
  auto nb = [](long long n, int t) { return (int)((n + t - 1) / t); };
  dim3 wblk(128);
  auto g2 = [](int M, int N) { return dim3((N + 63) / 64, (M + 63) / 64); };
  auto zfill = [&](u16* p, size_t elems) {   // bf16 zero-fill via u32 stores
    long long n = (long long)(elems / 2);
    k_fill_u32<<<nb(n, T), T, 0, stream>>>((u32*)p, 0u, n);
  };

  // ---- zero-fill padded bf16 operand buffers ----
  zfill(xb,     (size_t)NN_P * KX);
  zfill(hb,     (size_t)NN_P * HD_P);
  zfill(gat_wt, (size_t)NW1 * KX);
  zfill(gcn_wt, (size_t)NW1 * HD_P);
  zfill(gcat,   (size_t)NG * KG1);
  zfill(gg,     (size_t)NG * KG2);
  zfill(fcg1wt, (size_t)NF1 * KG1);
  zfill(fcg2wt, (size_t)128 * KG2);
  zfill(c1wt,   (size_t)64 * 2048);
  zfill(c2wt,   (size_t)64 * 512);
  zfill(c3wt,   (size_t)128 * 1024);
  zfill(xt0,    (size_t)NG * LROW * 128);
  zfill(y1,     (size_t)NG * LROW * C1);
  zfill(y2,     (size_t)NG * LROW * C2);

  // ---- weight prep ----
  k_copy_pad_bf16<<<nb((long long)NN * FXD, T), T, 0, stream>>>(x, xb, NN, FXD, KX);
  k_transpose_bf16<<<nb(FXD * HDIM, T), T, 0, stream>>>(gat_w, gat_wt, FXD, HDIM, KX);
  k_transpose_bf16<<<nb(HDIM * HDIM, T), T, 0, stream>>>(gcn_w, gcn_wt, HDIM, HDIM, HD_P);
  k_transpose_bf16<<<nb(1560 * 1500, T), T, 0, stream>>>(fcg1_w, fcg1wt, 1560, 1500, KG1);
  k_transpose_bf16<<<nb(1500 * 128, T), T, 0, stream>>>(fcg2_w, fcg2wt, 1500, 128, KG2);
  k_transpose_bf16<<<nb(96 * 128, T), T, 0, stream>>>(fcxt_w, fcxtwt, 96, 128, 96);
  k_transpose_bf16<<<nb(256 * 1024, T), T, 0, stream>>>(fc1_w, fc1wt, 256, 1024, 256);
  k_transpose_bf16<<<nb(1024 * 512, T), T, 0, stream>>>(fc2_w, fc2wt, 1024, 512, 1024);
  k_conv_wreorder<<<nb(C1 * 128 * 16, T), T, 0, stream>>>(c1_w, c1wt, C1, 128);
  k_conv_wreorder<<<nb(C2 * C1 * 16, T), T, 0, stream>>>(c2_w, c2wt, C2, C1);
  k_conv_wreorder<<<nb(C3 * C2 * 16, T), T, 0, stream>>>(c3_w, c3wt, C3, C2);
  k_bn_fold<<<1, C1, 0, stream>>>(c1_b, bn1g, bn1b, bn1m, bn1v, s1, t1, C1);
  k_bn_fold<<<1, C2, 0, stream>>>(c2_b, bn2g, bn2b, bn2m, bn2v, s2, t2, C2);
  k_bn_fold<<<1, C3, 0, stream>>>(c3_b, bn3g, bn3b, bn3m, bn3v, s3, t3, C3);
  k_bn_fold<<<1, 128, 0, stream>>>(fcxt_b, bnfg, bnfb, bnfm, bnfv, sf, tf, 128);

  // ---- GAT: h0 = x @ gat_w (WMMA) ----
  k_gemm_bf16<<<g2(NN, HDIM), wblk, 0, stream>>>(xb, gat_wt, bufP, nullptr,
      NN, HDIM, KX, HDIM, nullptr, nullptr, 0);
  k_gat_scores<<<nb((long long)NN * HEADS, T), T, 0, stream>>>(bufP, asrc, adst, as_, ad_, NN);
  k_fill_f32<<<nb((long long)NN * HEADS, T), T, 0, stream>>>(mmax, -1e30f, NN * HEADS);
  k_fill_f32<<<nb((long long)NN * HEADS, T), T, 0, stream>>>(zsum, 0.0f, NN * HEADS);
  k_fill_f32<<<nb((long long)NN * HDIM, T), T, 0, stream>>>(bufQ, 0.0f, NN * HDIM);
  k_fill_f32<<<nb(NN, T), T, 0, stream>>>(deg, 0.0f, NN);
  k_edge_max<<<nb((long long)ET * HEADS, T), T, 0, stream>>>(ei, as_, ad_, mmax, NE, ET);
  k_edge_p<<<nb((long long)ET * HEADS, T), T, 0, stream>>>(ei, as_, ad_, mmax, pbuf, zsum, NE, ET);
  k_deg<<<nb(ET, T), T, 0, stream>>>(ei, deg, NE, ET);
  k_edge_aggr_gat<<<ET, 256, 0, stream>>>(ei, bufP, pbuf, zsum, bufQ, NE);
  k_gat_finish<<<nb((long long)NN * HDIM, T), T, 0, stream>>>(bufQ, gat_b, hb, NN, HD_P);

  // ---- GCN: hw = h @ gcn_w (WMMA), symmetric-norm aggregation ----
  k_gemm_bf16<<<g2(NN, HDIM), wblk, 0, stream>>>(hb, gcn_wt, bufQ, nullptr,
      NN, HDIM, HD_P, HDIM, nullptr, nullptr, 0);
  k_fill_f32<<<nb((long long)NN * HDIM, T), T, 0, stream>>>(bufP, 0.0f, NN * HDIM);
  k_edge_aggr_gcn<<<ET, 256, 0, stream>>>(ei, bufQ, deg, bufP, NE);
  k_gcn_finish<<<nb((long long)NN * HDIM, T), T, 0, stream>>>(bufP, gcn_b, NN * HDIM);

  // ---- pooling + graph FCs ----
  k_fill_f32<<<nb((long long)NG * HDIM, T), T, 0, stream>>>(gmx, -1e30f, NG * HDIM);
  k_fill_f32<<<nb((long long)NG * HDIM, T), T, 0, stream>>>(gsm, 0.0f, NG * HDIM);
  k_fill_f32<<<nb(NG, T), T, 0, stream>>>(cnt, 0.0f, NG);
  k_pool<<<nb((long long)NN * HDIM, T), T, 0, stream>>>(bufP, batch, gmx, gsm, cnt, NN);
  k_pool_finish<<<nb((long long)NG * HDIM, T), T, 0, stream>>>(gmx, gsm, cnt, gcat, NG, KG1);
  k_gemm_bf16<<<g2(NG, 1500), wblk, 0, stream>>>(gcat, fcg1wt, nullptr, gg,
      NG, 1500, KG1, KG2, nullptr, fcg1_b, 1);
  k_gemm_bf16<<<g2(NG, 128), wblk, 0, stream>>>(gg, fcg2wt, nullptr, headin,
      NG, 128, KG2, 256, nullptr, fcg2_b, 0);

  // ---- protein CNN branch (WMMA implicit-GEMM convs, fused BN+ReLU) ----
  k_embed<<<nb((long long)NG * SEQ * 128, T), T, 0, stream>>>(target, emb, xt0,
      NG * SEQ * 128, LROW);
  k_conv_wmma<<<dim3((C1 + 63) / 64, (L1 + 63) / 64, NG), wblk, 0, stream>>>(
      xt0, c1wt, y1, L1, 7, C1, (long long)LROW * 128, (long long)LROW * C1, s1, t1);
  k_conv_wmma<<<dim3((C2 + 63) / 64, (L2 + 63) / 64, NG), wblk, 0, stream>>>(
      y1, c2wt, y2, L2, 5, C2, (long long)LROW * C1, (long long)LROW * C2, s2, t2);
  k_conv_wmma<<<dim3((C3 + 63) / 64, (L3 + 63) / 64, NG), wblk, 0, stream>>>(
      y2, c3wt, y3, L3, 6, C3, (long long)LROW * C2, (long long)Y3R * C3, s3, t3);
  k_maxpool_conv<<<nb(NG * C3, T), T, 0, stream>>>(y3, pooled, L3, Y3R, C3, NG);
  k_gemm_bf16<<<g2(NG, 128), wblk, 0, stream>>>(pooled, fcxtwt, nullptr, headin + 128,
      NG, 128, 96, 256, sf, tf, 1);

  // ---- head ----
  k_gemm_bf16<<<g2(NG, 1024), wblk, 0, stream>>>(headin, fc1wt, nullptr, a1,
      NG, 1024, 256, 1024, nullptr, fc1_b, 1);
  k_gemm_bf16<<<g2(NG, 512), wblk, 0, stream>>>(a1, fc2wt, nullptr, a2,
      NG, 512, 1024, 512, nullptr, fc2_b, 1);
  k_out_dot<<<nb(NG, T), T, 0, stream>>>(a2, out_w, out_b, (float*)d_out, NG);
}
